// MV_LSTM_38233798869098
// MI455X (gfx1250) — matrix operand order repaired and checked by reading, not compiled, not run on hardware
//
#include <hip/hip_runtime.h>

// ---------------------------------------------------------------------------
// MV_LSTM on MI455X (gfx1250): 2-layer LSTM + linear head.
//  k_xg0  : wide mixed-precision GEMM (f16 WMMA, f32 acc) for x @ w_ih0^T + b
//  k_lstm1: sequential recurrence, 16 batch rows per wave, 8 WMMAs/step
//  k_lstm2: recurrence with fused input projection (2 chained WMMAs/tile)
//           + on-the-fly final linear reduction
// Register-pressure control: `#pragma unroll 1` on fragment loops so LICM
// cannot hoist 64+ LDS fragments into registers and spill to scratch.
// ---------------------------------------------------------------------------

typedef __attribute__((ext_vector_type(16))) _Float16 v16h;
typedef __attribute__((ext_vector_type(8)))  _Float16 v8h;
typedef __attribute__((ext_vector_type(8)))  float    v8f;
typedef __attribute__((ext_vector_type(4)))  float    v4f;

#define BATCH 1024
#define SEQ   128
#define NF    256
#define NH    30
#define NG    120   /* 4*NH */

__device__ __forceinline__ float sigmoidf_(float x) {
  return 1.0f / (1.0f + __expf(-x));
}

// ===========================================================================
// Kernel 1: xg0[b*T + t][n] = sum_k x[b,t,k] * w_ih0[n,k] + (b_ih0[n]+b_hh0[n])
// M = B*T = 131072 (8192 tiles of 16), N = 120 -> 128, K = 256 (8 k-steps).
// 128 blocks x 8 waves, 8 M-tiles per wave.
// ===========================================================================
__global__ __launch_bounds__(256) void k_xg0(
    const float* __restrict__ x, const float* __restrict__ w_ih0,
    const float* __restrict__ b_ih0, const float* __restrict__ b_hh0,
    float* __restrict__ xg0)
{
  // 64 B-fragments (kt=0..7, nt=0..7), each 32x16 f16 in WMMA B layout:
  // lane%16 = N column, element e = K row (table order per lane half). 64 KB LDS.
  __shared__ v16h Bf[64 * 32];
  const int tid  = threadIdx.x;
  const int lane = tid & 31;
  const int wv   = tid >> 5;

  { // stage weights: wave wv stages kt = wv, all 8 nt
    const int kt = wv;
    const int kb = kt * 32 + (lane >> 4) * 8;
    for (int nt = 0; nt < 8; ++nt) {
      v16h tmp;
      #pragma unroll
      for (int e = 0; e < 16; ++e) {
        const int n = nt * 16 + (lane & 15);
        const int k = kb + (e < 8 ? e : 8 + e);
        tmp[e] = (_Float16)((n < NG) ? w_ih0[n * NF + k] : 0.0f);
      }
      Bf[(kt * 8 + nt) * 32 + lane] = tmp;
    }
  }
  __syncthreads();

  float bias[8];
  #pragma unroll
  for (int t = 0; t < 8; ++t) {
    const int n = t * 16 + (lane & 15);
    bias[t] = (n < NG) ? (b_ih0[n] + b_hh0[n]) : 0.0f;
  }

  const int hi  = lane >> 4;          // A-fragment half select
  const int kb0 = hi * 8;             // elements 0..7  -> K = kb0..kb0+7
  const int kb1 = 16 + hi * 8;        // elements 8..15 -> K = kb1..kb1+7
  const int gw  = blockIdx.x * 8 + wv;

  #pragma unroll 1
  for (int it = 0; it < 8; ++it) {
    const long m0 = ((long)gw * 8 + it) * 16;
    const float* arow = x + (m0 + (lane & 15)) * NF;
    v8f acc[8];
    #pragma unroll
    for (int t = 0; t < 8; ++t) acc[t] = (v8f){0,0,0,0,0,0,0,0};

    #pragma unroll 1            // keep <=8 B fragments live; no LICM blow-up
    for (int kt = 0; kt < 8; ++kt) {
      // A fragment: 16x32 f16 (converted from f32), per ISA 7.12.2 layout
      v4f a0 = *(const v4f*)(arow + kt * 32 + kb0);
      v4f a1 = *(const v4f*)(arow + kt * 32 + kb0 + 4);
      v4f a2 = *(const v4f*)(arow + kt * 32 + kb1);
      v4f a3 = *(const v4f*)(arow + kt * 32 + kb1 + 4);
      v16h a;
      #pragma unroll
      for (int j = 0; j < 4; ++j) {
        a[j]      = (_Float16)a0[j];
        a[4 + j]  = (_Float16)a1[j];
        a[8 + j]  = (_Float16)a2[j];
        a[12 + j] = (_Float16)a3[j];
      }
      #pragma unroll
      for (int t = 0; t < 8; ++t) {
        acc[t] = __builtin_amdgcn_wmma_f32_16x16x32_f16(
            false, a, false, Bf[(kt * 8 + t) * 32 + lane],
            (short)0, acc[t], false, false);
      }
    }
    // epilogue: D layout lane=(N, half), VGPR=M; add bias, store f32
    float* orow = xg0 + m0 * 128;
    #pragma unroll
    for (int t = 0; t < 8; ++t) {
      const int n = t * 16 + (lane & 15);
      #pragma unroll
      for (int v = 0; v < 8; ++v) {
        orow[(long)(v + 8 * hi) * 128 + n] = acc[t][v] + bias[t];
      }
    }
  }
}

// ===========================================================================
// Kernel 2: layer-0 recurrence. 16 blocks x 4 waves; wave owns 16 batch rows.
// Per step: gates[16,128] = h[16,32pad] @ w_hh0^T (one K=32 WMMA per n-tile),
// add xg0, cell update, repack h into A-fragment layout, emit h1 as f16.
// ===========================================================================
__global__ __launch_bounds__(128) void k_lstm1(
    const float* __restrict__ xg0, const float* __restrict__ w_hh0,
    _Float16* __restrict__ h1g)
{
  __shared__ v16h Wf[8 * 32];                    // w_hh0 B fragments (8 KB)
  __shared__ __align__(32) _Float16 Hb[4][512];  // per-wave h state [m][k] (4 KB)
  __shared__ float Gl[4][128 * 16];              // per-wave gates [n][m] (32 KB)
  const int tid  = threadIdx.x;
  const int lane = tid & 31;
  const int wv   = tid >> 5;

  for (int q = 0; q < 2; ++q) {      // each wave stages 2 of 8 fragments
    const int nt = wv * 2 + q;
    v16h tmp;
    #pragma unroll
    for (int e = 0; e < 16; ++e) {
      const int n = nt * 16 + (lane & 15);
      const int k = (lane >> 4) * 8 + (e < 8 ? e : 8 + e);
      tmp[e] = (_Float16)((k < NH && n < NG) ? w_hh0[n * NH + k] : 0.0f);
    }
    Wf[nt * 32 + lane] = tmp;
  }
  #pragma unroll
  for (int j = 0; j < 16; ++j) Hb[wv][lane * 16 + j] = (_Float16)0.0f;
  __syncthreads();

  const int b0  = (blockIdx.x * 4 + wv) * 16;
  const int hi  = lane >> 4;
  const int kb0 = hi * 8;
  const int kb1 = 16 + hi * 8;
  float c[16];
  #pragma unroll
  for (int m = 0; m < 16; ++m) c[m] = 0.0f;

  #pragma unroll 1
  for (int t = 0; t < SEQ; ++t) {
    // A fragment for h (16x32, K padded 30->32 with zeros)
    const _Float16* hr = &Hb[wv][(lane & 15) * 32];
    v8h lo = *(const v8h*)(hr + kb0);
    v8h hh = *(const v8h*)(hr + kb1);
    v16h a = __builtin_shufflevector(lo, hh, 0,1,2,3,4,5,6,7,8,9,10,11,12,13,14,15);

    #pragma unroll 1            // one weight fragment live at a time
    for (int nt = 0; nt < 8; ++nt) {
      v8f acc = (v8f){0,0,0,0,0,0,0,0};
      acc = __builtin_amdgcn_wmma_f32_16x16x32_f16(
          false, a, false, Wf[nt * 32 + lane], (short)0, acc, false, false);
      const int n = nt * 16 + (lane & 15);
      #pragma unroll
      for (int v = 0; v < 8; ++v) Gl[wv][n * 16 + v + 8 * hi] = acc[v];
    }
    __syncthreads();

    if (lane < NH) {               // lane = hidden index h
      const int h = lane;
      #pragma unroll 4
      for (int m = 0; m < 16; ++m) {
        const size_t xb = ((size_t)(b0 + m) * SEQ + t) * 128;
        float iv = Gl[wv][(h)          * 16 + m] + xg0[xb + h];
        float fv = Gl[wv][(NH + h)     * 16 + m] + xg0[xb + NH + h];
        float gv = Gl[wv][(2 * NH + h) * 16 + m] + xg0[xb + 2 * NH + h];
        float ov = Gl[wv][(3 * NH + h) * 16 + m] + xg0[xb + 3 * NH + h];
        iv = sigmoidf_(iv); fv = sigmoidf_(fv);
        gv = tanhf(gv);     ov = sigmoidf_(ov);
        const float cn = fv * c[m] + iv * gv;
        c[m] = cn;
        const _Float16 h16 = (_Float16)(ov * tanhf(cn));
        Hb[wv][m * 32 + h] = h16;                               // next-step A
        h1g[((size_t)(b0 + m) * SEQ + t) * 32 + h] = h16;       // layer-2 input
      }
    }
    __syncthreads();
  }
}

// ===========================================================================
// Kernel 3: layer-1 recurrence with fused input projection + linear head.
// gates = h1_t @ w_ih1^T + h2 @ w_hh1^T + bias  (two chained WMMAs per tile)
// out[b] = sum_{t,h} h2[b,t,h] * w_lin[t*30+h] + b_lin
// ===========================================================================
__global__ __launch_bounds__(128) void k_lstm2(
    const _Float16* __restrict__ h1g,
    const float* __restrict__ w_ih1, const float* __restrict__ w_hh1,
    const float* __restrict__ b_ih1, const float* __restrict__ b_hh1,
    const float* __restrict__ w_lin, const float* __restrict__ b_lin,
    float* __restrict__ out)
{
  __shared__ v16h WI[8 * 32];                    // w_ih1 B fragments (8 KB)
  __shared__ v16h WH[8 * 32];                    // w_hh1 B fragments (8 KB)
  __shared__ __align__(32) _Float16 Hb[4][512];  // h2 state (4 KB)
  __shared__ float Gl[4][128 * 16];              // gates / reduction (32 KB)
  const int tid  = threadIdx.x;
  const int lane = tid & 31;
  const int wv   = tid >> 5;

  for (int q = 0; q < 2; ++q) {
    const int nt = wv * 2 + q;
    v16h ti, th;
    #pragma unroll
    for (int e = 0; e < 16; ++e) {
      const int n = nt * 16 + (lane & 15);
      const int k = (lane >> 4) * 8 + (e < 8 ? e : 8 + e);
      const bool ok = (k < NH) && (n < NG);
      ti[e] = (_Float16)(ok ? w_ih1[n * NH + k] : 0.0f);
      th[e] = (_Float16)(ok ? w_hh1[n * NH + k] : 0.0f);
    }
    WI[nt * 32 + lane] = ti;
    WH[nt * 32 + lane] = th;
  }
  #pragma unroll
  for (int j = 0; j < 16; ++j) Hb[wv][lane * 16 + j] = (_Float16)0.0f;
  __syncthreads();

  const int b0  = (blockIdx.x * 4 + wv) * 16;
  const int hi  = lane >> 4;
  const int kb0 = hi * 8;
  const int kb1 = 16 + hi * 8;
  float c[16], lin[16];
  #pragma unroll
  for (int m = 0; m < 16; ++m) { c[m] = 0.0f; lin[m] = 0.0f; }

  float b4[4] = {0.0f, 0.0f, 0.0f, 0.0f};
  if (lane < NH) {
    b4[0] = b_ih1[lane]          + b_hh1[lane];
    b4[1] = b_ih1[NH + lane]     + b_hh1[NH + lane];
    b4[2] = b_ih1[2 * NH + lane] + b_hh1[2 * NH + lane];
    b4[3] = b_ih1[3 * NH + lane] + b_hh1[3 * NH + lane];
  }

  #pragma unroll 1
  for (int t = 0; t < SEQ; ++t) {
    // A1: h1_t from global (already f16, fragment-friendly [b][t][32] layout)
    const _Float16* xr = h1g + ((size_t)(b0 + (lane & 15)) * SEQ + t) * 32;
    v8h xlo = *(const v8h*)(xr + kb0);
    v8h xhi = *(const v8h*)(xr + kb1);
    v16h a1 = __builtin_shufflevector(xlo, xhi, 0,1,2,3,4,5,6,7,8,9,10,11,12,13,14,15);
    // A2: h2 state from LDS
    const _Float16* hr = &Hb[wv][(lane & 15) * 32];
    v8h hlo = *(const v8h*)(hr + kb0);
    v8h hhi = *(const v8h*)(hr + kb1);
    v16h a2 = __builtin_shufflevector(hlo, hhi, 0,1,2,3,4,5,6,7,8,9,10,11,12,13,14,15);

    #pragma unroll 1            // two weight fragments live at a time
    for (int nt = 0; nt < 8; ++nt) {
      v8f acc = (v8f){0,0,0,0,0,0,0,0};
      acc = __builtin_amdgcn_wmma_f32_16x16x32_f16(
          false, a1, false, WI[nt * 32 + lane], (short)0, acc, false, false);
      acc = __builtin_amdgcn_wmma_f32_16x16x32_f16(
          false, a2, false, WH[nt * 32 + lane], (short)0, acc, false, false);
      const int n = nt * 16 + (lane & 15);
      #pragma unroll
      for (int v = 0; v < 8; ++v) Gl[wv][n * 16 + v + 8 * hi] = acc[v];
    }
    __syncthreads();

    if (lane < NH) {
      const int h = lane;
      const float wl = w_lin[t * NH + h];
      #pragma unroll 4
      for (int m = 0; m < 16; ++m) {
        float iv = Gl[wv][(h)          * 16 + m] + b4[0];
        float fv = Gl[wv][(NH + h)     * 16 + m] + b4[1];
        float gv = Gl[wv][(2 * NH + h) * 16 + m] + b4[2];
        float ov = Gl[wv][(3 * NH + h) * 16 + m] + b4[3];
        iv = sigmoidf_(iv); fv = sigmoidf_(fv);
        gv = tanhf(gv);     ov = sigmoidf_(ov);
        const float cn = fv * c[m] + iv * gv;
        c[m] = cn;
        const float hn = ov * tanhf(cn);
        lin[m] += hn * wl;                        // linear head, fused
        Hb[wv][m * 32 + h] = (_Float16)hn;
      }
    }
    __syncthreads();
  }

  // reduce lin over hidden lanes -> 16 outputs per wave (reuse Gl as scratch)
  float* Rb = &Gl[wv][0];
  #pragma unroll
  for (int m = 0; m < 16; ++m) Rb[lane * 16 + m] = lin[m];  // lanes>=30 wrote 0
  __syncthreads();
  if (lane < 16) {
    float s = b_lin[0];
    for (int l = 0; l < 32; ++l) s += Rb[l * 16 + lane];
    out[b0 + lane] = s;
  }
}

// ===========================================================================
extern "C" void kernel_launch(void* const* d_in, const int* in_sizes, int n_in,
                              void* d_out, int out_size, void* d_ws, size_t ws_size,
                              hipStream_t stream) {
  (void)in_sizes; (void)n_in; (void)out_size; (void)ws_size;
  const float* x     = (const float*)d_in[0];
  const float* w_ih0 = (const float*)d_in[1];
  const float* w_hh0 = (const float*)d_in[2];
  const float* b_ih0 = (const float*)d_in[3];
  const float* b_hh0 = (const float*)d_in[4];
  const float* w_ih1 = (const float*)d_in[5];
  const float* w_hh1 = (const float*)d_in[6];
  const float* b_ih1 = (const float*)d_in[7];
  const float* b_hh1 = (const float*)d_in[8];
  const float* w_lin = (const float*)d_in[9];
  const float* b_lin = (const float*)d_in[10];
  float* out = (float*)d_out;

  // workspace: xg0 (B*T x 128 f32, 64 MB) then h1 (B*T x 32 f16, 8 MB)
  float*    xg0 = (float*)d_ws;
  _Float16* h1g = (_Float16*)((char*)d_ws +
                              (size_t)BATCH * SEQ * 128 * sizeof(float));

  k_xg0  <<<128, 256, 0, stream>>>(x, w_ih0, b_ih0, b_hh0, xg0);
  k_lstm1<<<16, 128, 0, stream>>>(xg0, w_hh0, h1g);
  k_lstm2<<<16, 128, 0, stream>>>(h1g, w_ih1, w_hh1, b_ih1, b_hh1,
                                  w_lin, b_lin, out);
}
